// Attention_46420006535531
// MI455X (gfx1250) — matrix operand order repaired and checked by reading, not compiled, not run on hardware
//
#include <hip/hip_runtime.h>
#include <hip/hip_bf16.h>
#include <math.h>

typedef _Float16 half_t;
typedef __attribute__((ext_vector_type(16))) _Float16 v16h;
typedef __attribute__((ext_vector_type(8)))  _Float16 v8h;
typedef __attribute__((ext_vector_type(8)))  float    v8f;

#define WMMA_F16(a, b, c) \
    __builtin_amdgcn_wmma_f32_16x16x32_f16(false, (a), false, (b), (short)0, (c), false, false)

// ---- constants -------------------------------------------------------------
static constexpr int B_    = 4;
static constexpr int N_    = 1024;
static constexpr int DIM   = 256;
static constexpr int HEADS = 8;
static constexpr int DHEAD = 64;
static constexpr int INNER = 512;   // HEADS*DHEAD
static constexpr int MROWS = B_ * N_;  // 4096

// ---- fragment load helpers -------------------------------------------------
// A fragment (16x32 f16): lane half h holds K in [16h,16h+7] and [16h+8,16h+15]
__device__ inline v16h ld_a_frag(const half_t* lo, const half_t* hi) {
    v16h a;
    v8h l = *(const v8h*)lo;
    v8h h = *(const v8h*)hi;
#pragma unroll
    for (int i = 0; i < 8; ++i) { a[i] = l[i]; a[i + 8] = h[i]; }
    return a;
}
// B fragment (32x16 f16): lane half h holds K in [16h, 16h+15], contiguous
__device__ inline v16h ld_b_frag(const half_t* p) {
    return *(const v16h*)p;
}
// A fragment from fp32 source (8 floats at p, 8 at p+8)
__device__ inline v16h cvt_a_frag(const float* p) {
    v16h a;
#pragma unroll
    for (int i = 0; i < 8; ++i) {
        a[i]     = (_Float16)p[i];
        a[i + 8] = (_Float16)p[8 + i];
    }
    return a;
}

// ---- kernel 1: convert & transpose weights to f16 --------------------------
// WqT/WkT/WvT/WgT: [512 col][256 k] f16 ; WoT: [256 col][512 k] f16
__global__ void cvt_weights_kernel(const float* __restrict__ Wq,
                                   const float* __restrict__ Wkv,
                                   const float* __restrict__ Wg,
                                   const float* __restrict__ Wo,
                                   half_t* __restrict__ WqT, half_t* __restrict__ WkT,
                                   half_t* __restrict__ WvT, half_t* __restrict__ WgT,
                                   half_t* __restrict__ WoT) {
    const int SZ = 512 * 256;
    int tid = blockIdx.x * 256 + threadIdx.x;
    if (tid < 4 * SZ) {
        int m = tid / SZ, rem = tid % SZ;
        int c = rem / 256, k = rem % 256;
        float v; half_t* dst;
        if (m == 0)      { v = Wq[k * 512 + c];         dst = WqT; }
        else if (m == 1) { v = Wkv[k * 1024 + c];       dst = WkT; }
        else if (m == 2) { v = Wkv[k * 1024 + 512 + c]; dst = WvT; }
        else             { v = Wg[k * 512 + c];         dst = WgT; }
        dst[c * 256 + k] = (half_t)v;
    } else if (tid < 5 * SZ) {
        int rem = tid - 4 * SZ;
        int c = rem / 512, k = rem % 512;
        WoT[c * 512 + k] = (half_t)Wo[k * 256 + c];
    }
}

// ---- kernel 2: fused QKV+G projection --------------------------------------
// x[4096,256] x W[256,2048]  (regions: q | k | v | g, each 512 cols)
// q -> qbuf[bh][n][d] (pre-scaled), k -> kbuf[bh][n][d], v -> vT[bh*64+d][n],
// g -> gbuf[row][512] = sigmoid(. + bg)
__global__ __launch_bounds__(32) void proj_kernel(
        const float* __restrict__ x,
        const half_t* __restrict__ WqT, const half_t* __restrict__ WkT,
        const half_t* __restrict__ WvT, const half_t* __restrict__ WgT,
        const float* __restrict__ bg,
        half_t* __restrict__ qbuf, half_t* __restrict__ kbuf,
        half_t* __restrict__ vT, float* __restrict__ gbuf) {
    const int ct   = blockIdx.x;          // 0..127 (col tile over 2048)
    const int mt   = blockIdx.y;          // 0..255
    const int lane = threadIdx.x;
    const int hl   = lane >> 4;           // lane half
    const int lr   = lane & 15;

    const int region = ct >> 5;           // 0..3
    const int cb     = (ct & 31) * 16;    // col base within region
    const half_t* WT = (region == 0) ? WqT : (region == 1) ? WkT
                     : (region == 2) ? WvT : WgT;
    const half_t* wrow = WT + (size_t)(cb + lr) * 256;
    const float*  xrow = x + (size_t)(mt * 16 + lr) * 256;

    v8f acc = {};
#pragma unroll
    for (int k0 = 0; k0 < 256; k0 += 32) {
        v16h a = cvt_a_frag(xrow + k0 + 16 * hl);
        v16h b = ld_b_frag(wrow + k0 + 16 * hl);
        acc = WMMA_F16(a, b, acc);
    }

    const int col = cb + lr;              // 0..511 within region
    const int h = col >> 6, d = col & 63;
#pragma unroll
    for (int j = 0; j < 8; ++j) {
        int r = mt * 16 + j + 8 * hl;     // global row in [0,4096)
        int bb = r >> 10, n = r & 1023;
        float v = acc[j];
        if (region == 0) {
            qbuf[((size_t)(bb * HEADS + h) * N_ + n) * DHEAD + d] = (half_t)(v * 0.125f);
        } else if (region == 1) {
            kbuf[((size_t)(bb * HEADS + h) * N_ + n) * DHEAD + d] = (half_t)v;
        } else if (region == 2) {
            vT[((size_t)(bb * HEADS + h) * DHEAD + d) * N_ + n] = (half_t)v;
        } else {
            float g = 1.0f / (1.0f + __expf(-(v + bg[col])));
            gbuf[(size_t)r * INNER + col] = g;
        }
    }
}

// ---- kernel 3: flash attention (one wave per (bh, 16-query tile)) ----------
__global__ __launch_bounds__(32) void attn_kernel(
        const half_t* __restrict__ qbuf, const half_t* __restrict__ kbuf,
        const half_t* __restrict__ vT,   const float* __restrict__ bias,
        const float* __restrict__ gbuf,  half_t* __restrict__ obuf) {
    __shared__ __align__(32) _Float16 pts[16 * 32];   // P tile, row-major 16x32

    const int qt   = blockIdx.x;          // 0..63
    const int bh   = blockIdx.y;          // 0..31
    const int lane = threadIdx.x;
    const int hl   = lane >> 4;
    const int lr   = lane & 15;
    const int qbase = qt * 16;

    // Q A-fragments (d = 0..31 and 32..63), already scaled by dhead^-0.5
    const half_t* qrow = qbuf + ((size_t)bh * N_ + qbase + lr) * DHEAD;
    v16h qA0 = ld_a_frag(qrow + 16 * hl,      qrow + 8 + 16 * hl);
    v16h qA1 = ld_a_frag(qrow + 32 + 16 * hl, qrow + 40 + 16 * hl);

    float m[8], l[8];
#pragma unroll
    for (int j = 0; j < 8; ++j) { m[j] = -1e30f; l[j] = 0.0f; }
    v8f o0 = {}, o1 = {}, o2 = {}, o3 = {};

    const float* bp = bias + ((size_t)bh * N_ + qbase) * N_;

    for (int kb = 0; kb < N_; kb += 32) {
        // --- S = Q @ K^T  (two 16-col tiles) --------------------------------
        const half_t* kp0 = kbuf + ((size_t)bh * N_ + kb + lr) * DHEAD + 16 * hl;
        const half_t* kp1 = kp0 + 16 * DHEAD;
        v8f s0 = {}, s1 = {};
        s0 = WMMA_F16(qA0, ld_b_frag(kp0),      s0);
        s0 = WMMA_F16(qA1, ld_b_frag(kp0 + 32), s0);
        s1 = WMMA_F16(qA0, ld_b_frag(kp1),      s1);
        s1 = WMMA_F16(qA1, ld_b_frag(kp1 + 32), s1);

        // prefetch next bias tile (speculative; silently dropped at the end)
        __builtin_prefetch(bp + (size_t)(lane & 15) * N_ + kb + 32 + hl * 16, 0, 1);

        // --- add bias (streamed non-temporally) -----------------------------
#pragma unroll
        for (int j = 0; j < 8; ++j) {
            int r = j + 8 * hl;
            s0[j] += __builtin_nontemporal_load(bp + (size_t)r * N_ + kb + lr);
            s1[j] += __builtin_nontemporal_load(bp + (size_t)r * N_ + kb + 16 + lr);
        }

        // --- online softmax update ------------------------------------------
        float alpha[8], p0[8], p1[8];
#pragma unroll
        for (int j = 0; j < 8; ++j) {
            float vmax = fmaxf(s0[j], s1[j]);
#pragma unroll
            for (int off = 1; off < 16; off <<= 1)
                vmax = fmaxf(vmax, __shfl_xor(vmax, off, 32));
            float mn = fmaxf(m[j], vmax);
            float al = __expf(m[j] - mn);
            float e0 = __expf(s0[j] - mn);
            float e1 = __expf(s1[j] - mn);
            float rs = e0 + e1;
#pragma unroll
            for (int off = 1; off < 16; off <<= 1)
                rs += __shfl_xor(rs, off, 32);
            l[j] = l[j] * al + rs;
            m[j] = mn;
            alpha[j] = al; p0[j] = e0; p1[j] = e1;
        }
#pragma unroll
        for (int j = 0; j < 8; ++j) {
            o0[j] *= alpha[j]; o1[j] *= alpha[j];
            o2[j] *= alpha[j]; o3[j] *= alpha[j];
        }

        // --- relayout P from C-layout to A-layout via LDS -------------------
#pragma unroll
        for (int j = 0; j < 8; ++j) {
            int r = j + 8 * hl;
            pts[r * 32 + lr]      = (half_t)p0[j];
            pts[r * 32 + 16 + lr] = (half_t)p1[j];
        }
        __syncthreads();   // single-wave WG: cheap; enforces DS ordering
        v16h pA = ld_a_frag(pts + lr * 32 + 16 * hl, pts + lr * 32 + 8 + 16 * hl);
        __syncthreads();

        // --- O += P @ V  (vT gives contiguous B-fragments) ------------------
        const half_t* vbase = vT + ((size_t)bh * DHEAD + lr) * N_ + kb + 16 * hl;
        o0 = WMMA_F16(pA, ld_b_frag(vbase),             o0);
        o1 = WMMA_F16(pA, ld_b_frag(vbase + 16 * N_),   o1);
        o2 = WMMA_F16(pA, ld_b_frag(vbase + 32 * N_),   o2);
        o3 = WMMA_F16(pA, ld_b_frag(vbase + 48 * N_),   o3);
    }

    // --- epilogue: normalize, apply gate, store f16 -------------------------
    const int bb = bh >> 3, h = bh & 7;
    float rl[8];
#pragma unroll
    for (int j = 0; j < 8; ++j) rl[j] = 1.0f / l[j];
#pragma unroll
    for (int dc = 0; dc < 4; ++dc) {
        v8f oo = (dc == 0) ? o0 : (dc == 1) ? o1 : (dc == 2) ? o2 : o3;
        int col = h * DHEAD + dc * 16 + lr;
#pragma unroll
        for (int j = 0; j < 8; ++j) {
            int n = qbase + j + 8 * hl;
            size_t grow = (size_t)bb * N_ + n;
            float g = gbuf[grow * INNER + col];
            obuf[grow * INNER + col] = (half_t)(oo[j] * rl[j] * g);
        }
    }
}

// ---- kernel 4: output projection  out = obuf @ Wo + bo ---------------------
__global__ __launch_bounds__(32) void out_kernel(
        const half_t* __restrict__ obuf, const half_t* __restrict__ WoT,
        const float* __restrict__ bo, float* __restrict__ out) {
    const int ct   = blockIdx.x;          // 0..15
    const int mt   = blockIdx.y;          // 0..255
    const int lane = threadIdx.x;
    const int hl   = lane >> 4;
    const int lr   = lane & 15;

    const half_t* arow = obuf + (size_t)(mt * 16 + lr) * INNER;
    const half_t* wrow = WoT + (size_t)(ct * 16 + lr) * INNER;

    v8f acc = {};
#pragma unroll
    for (int k0 = 0; k0 < INNER; k0 += 32) {
        v16h a = ld_a_frag(arow + k0 + 16 * hl, arow + k0 + 8 + 16 * hl);
        v16h b = ld_b_frag(wrow + k0 + 16 * hl);
        acc = WMMA_F16(a, b, acc);
    }
    float bv = bo[ct * 16 + lr];
#pragma unroll
    for (int j = 0; j < 8; ++j) {
        int r = mt * 16 + j + 8 * hl;
        out[(size_t)r * DIM + ct * 16 + lr] = acc[j] + bv;
    }
}

// ---- launcher --------------------------------------------------------------
extern "C" void kernel_launch(void* const* d_in, const int* in_sizes, int n_in,
                              void* d_out, int out_size, void* d_ws, size_t ws_size,
                              hipStream_t stream) {
    const float* x    = (const float*)d_in[0];
    const float* bias = (const float*)d_in[1];
    const float* Wq   = (const float*)d_in[2];
    const float* Wkv  = (const float*)d_in[3];
    const float* Wo   = (const float*)d_in[4];
    const float* bo   = (const float*)d_in[5];
    const float* Wg   = (const float*)d_in[6];
    const float* bg   = (const float*)d_in[7];
    float* out = (float*)d_out;

    // workspace carve-up (256B aligned)
    char* ws = (char*)d_ws;
    size_t off = 0;
    auto carve = [&](size_t bytes) {
        char* p = ws + off;
        off = (off + bytes + 255) & ~(size_t)255;
        return p;
    };
    half_t* qbuf = (half_t*)carve((size_t)B_ * HEADS * N_ * DHEAD * 2); // 4 MB
    half_t* kbuf = (half_t*)carve((size_t)B_ * HEADS * N_ * DHEAD * 2); // 4 MB
    half_t* vT   = (half_t*)carve((size_t)B_ * HEADS * N_ * DHEAD * 2); // 4 MB
    float*  gbuf = (float*) carve((size_t)MROWS * INNER * 4);           // 8 MB
    half_t* obuf = (half_t*)carve((size_t)MROWS * INNER * 2);           // 4 MB
    half_t* WqT  = (half_t*)carve((size_t)512 * 256 * 2);
    half_t* WkT  = (half_t*)carve((size_t)512 * 256 * 2);
    half_t* WvT  = (half_t*)carve((size_t)512 * 256 * 2);
    half_t* WgT  = (half_t*)carve((size_t)512 * 256 * 2);
    half_t* WoT  = (half_t*)carve((size_t)256 * 512 * 2);

    // 1) weight convert/transpose to f16
    cvt_weights_kernel<<<2560, 256, 0, stream>>>(Wq, Wkv, Wg, Wo,
                                                 WqT, WkT, WvT, WgT, WoT);
    // 2) fused QKV+G projection
    proj_kernel<<<dim3(128, 256), 32, 0, stream>>>(x, WqT, WkT, WvT, WgT, bg,
                                                   qbuf, kbuf, vT, gbuf);
    // 3) flash attention + gating
    attn_kernel<<<dim3(64, 32), 32, 0, stream>>>(qbuf, kbuf, vT, bias, gbuf, obuf);
    // 4) output projection
    out_kernel<<<dim3(16, 256), 32, 0, stream>>>(obuf, WoT, bo, out);
}
